// Model_53506702573700
// MI455X (gfx1250) — compile-verified, matrix operand order checked by reading, not run on hardware
//
#include <hip/hip_runtime.h>
#include <cmath>

typedef __attribute__((ext_vector_type(16))) _Float16 v16h;
typedef __attribute__((ext_vector_type(8)))  float    v8f;

#define NX    48
#define NVOX  (48*48*48)          // 110592 voxels per batch image
#define NB    2
#define CPAD  64                  // padded channel count (f16 activations / weights)
#define NTAP  125                 // 5x5x5 conv taps
#define CENTER_TAP 62             // (2,2,2)

// ---------------------------------------------------------------------------
// Kernel builder: materialize dense f16 conv kernel W_T[tap][co][ci]
// (column-major in channels so WMMA B fragments are contiguous per lane).
// Faithful to build_kernel(): tensor-product paths (EMB x tw x SH x CG) at all
// taps, overridden by the self-connection Linear at the center tap.
// ---------------------------------------------------------------------------
__device__ __forceinline__ float cgv(int l1, int l2, int lo, int i, int j, int k) {
    if (l1 == 0 && l2 == 0 && lo == 0) return 1.0f;
    if (l1 == 0 && l2 == 1 && lo == 1) return (j == k) ? 1.0f : 0.0f;
    if (l1 == 1 && l2 == 0 && lo == 1) return (i == k) ? 1.0f : 0.0f;
    if (l1 == 1 && l2 == 1 && lo == 0) return (i == j) ? 0.57735026919f : 0.0f;
    // (1,1,1): levi-civita / sqrt(2)
    int e = 0;
    if ((i==0&&j==1&&k==2)||(i==1&&j==2&&k==0)||(i==2&&j==0&&k==1)) e = 1;
    else if ((i==1&&j==0&&k==2)||(i==2&&j==1&&k==0)||(i==0&&j==2&&k==1)) e = -1;
    return (float)e * 0.70710678119f;
}

__global__ void build_weights(int layer, const float* __restrict__ tw,
                              const float* __restrict__ lw,
                              _Float16* __restrict__ Wt) {
    const int tap = blockIdx.x;       // 0..124
    const int co  = threadIdx.x;      // 0..63

    // ---- irreps metadata for this layer ----
    int in_m[4], in_l[4], in_p[4], nin;
    int out_m[5], out_l[5], out_p[5], nout;
    if (layer == 0) {                                   // IR_A
        nin = 1; in_m[0]=1; in_l[0]=0; in_p[0]=1;
    } else {                                            // IR_C
        nin = 4;
        in_m[0]=10; in_l[0]=0; in_p[0]= 1;
        in_m[1]=10; in_l[1]=0; in_p[1]=-1;
        in_m[2]= 5; in_l[2]=1; in_p[2]= 1;
        in_m[3]= 5; in_l[3]=1; in_p[3]=-1;
    }
    if (layer < 4) {                                    // IR_B
        nout = 5;
        out_m[0]=10; out_l[0]=0; out_p[0]= 1;
        out_m[1]=10; out_l[1]=0; out_p[1]=-1;
        out_m[2]=10; out_l[2]=0; out_p[2]= 1;
        out_m[3]= 5; out_l[3]=1; out_p[3]= 1;
        out_m[4]= 5; out_l[4]=1; out_p[4]=-1;
    } else {                                            // IR_D
        nout = 2;
        out_m[0]=1; out_l[0]=0; out_p[0]=-1;
        out_m[1]=6; out_l[1]=0; out_p[1]= 1;
    }
    int sin_s[5], sout_s[6];
    int acc = 0;
    for (int i = 0; i < nin; ++i) { sin_s[i] = acc; acc += in_m[i]*(2*in_l[i]+1); }
    const int Cin = acc;
    acc = 0;
    for (int j = 0; j < nout; ++j) { sout_s[j] = acc; acc += out_m[j]*(2*out_l[j]+1); }
    const int Cout = acc;

    float val[64];
    #pragma unroll
    for (int c = 0; c < 64; ++c) val[c] = 0.0f;

    if (tap == CENTER_TAP) {
        // self-connection Linear (overwrites TP at center; EMB is 0 there anyway)
        int off = 0;
        for (int i = 0; i < nin; ++i)
            for (int j = 0; j < nout; ++j)
                if (in_l[i] == out_l[j] && in_p[i] == out_p[j]) {
                    const int m1 = in_m[i], l = in_l[i], mo = out_m[j];
                    const int so = sout_s[j], dim = 2*l+1;
                    if (co >= so && co < so + mo*dim) {
                        const int wch = (co - so) / dim, k = (co - so) % dim;
                        const float alpha = rsqrtf((float)m1);
                        for (int u = 0; u < m1; ++u)
                            val[sin_s[i] + u*dim + k] += alpha * lw[off + u*mo + wch];
                    }
                    off += m1 * mo;
                }
    } else {
        // lattice point, radial basis (smooth_finite, R=2), SH (0e+1o)
        const float cx = (float)(tap/25) - 2.0f;
        const float cy = (float)((tap/5)%5) - 2.0f;
        const float cz = (float)(tap%5) - 2.0f;
        const float d  = sqrtf(cx*cx + cy*cy + cz*cz);
        float emb[2];
        const float stepv = 2.5f / 3.0f;
        #pragma unroll
        for (int r = 0; r < 2; ++r) {
            const float diff = (d - (float)(r+1)*stepv) / stepv;
            const float t1 = diff + 1.0f, t2 = 1.0f - diff;
            const float s1 = (t1 > 0.0f) ? expf(-1.0f/t1) : 0.0f;
            const float s2 = (t2 > 0.0f) ? expf(-1.0f/t2) : 0.0f;
            emb[r] = 1.14136f * expf(2.0f) * s1 * s2;
        }
        const float inv = 1.0f / fmaxf(d, 1e-9f);
        float sh[4];
        sh[0] = 1.0f;
        sh[1] = 1.73205080757f * cy * inv;   // e3nn order (y,z,x)
        sh[2] = 1.73205080757f * cz * inv;
        sh[3] = 1.73205080757f * cx * inv;

        int off = 0;
        for (int i1 = 0; i1 < nin; ++i1)
            for (int i2 = 0; i2 < 2; ++i2)
                for (int io = 0; io < nout; ++io) {
                    const int l1 = in_l[i1], p1 = in_p[i1];
                    const int l2 = (i2 == 0) ? 0 : 1, p2 = (i2 == 0) ? 1 : -1;
                    const int lo = out_l[io], po = out_p[io];
                    const int lmin = (l1 > l2) ? (l1 - l2) : (l2 - l1);
                    if (po == p1*p2 && lo >= lmin && lo <= l1 + l2) {
                        const int m1 = in_m[i1], mo = out_m[io];
                        const int so = sout_s[io], dimo = 2*lo+1, dimi = 2*l1+1;
                        if (co >= so && co < so + mo*dimo) {
                            const int wch = (co - so) / dimo, k = (co - so) % dimo;
                            const float alpha = rsqrtf((float)(m1 * 2));
                            for (int i = 0; i < dimi; ++i) {
                                float s = 0.0f;
                                for (int j = 0; j < 2*l2+1; ++j)
                                    s += sh[((i2 == 0) ? 0 : 1) + j] * cgv(l1, l2, lo, i, j, k);
                                if (s != 0.0f) {
                                    for (int u = 0; u < m1; ++u) {
                                        const float wsum =
                                            emb[0] * tw[off + (0*m1 + u)*mo + wch] +
                                            emb[1] * tw[off + (1*m1 + u)*mo + wch];
                                        val[sin_s[i1] + u*dimi + i] += alpha * wsum * s;
                                    }
                                }
                            }
                        }
                        off += 2 * m1 * mo;
                    }
                }
    }

    _Float16* dst = Wt + ((long)tap*64 + co)*64;
    const bool co_ok = (co < Cout);
    for (int ci = 0; ci < 64; ++ci)
        dst[ci] = (_Float16)((co_ok && ci < Cin) ? val[ci] : 0.0f);
}

// ---------------------------------------------------------------------------
// Implicit-GEMM 5x5x5 conv via v_wmma_f32_16x16x32_f16.
// A: f16 activations [NB*NVOX][64] (channel-last). W: f16 [125][co=64][ci=64].
// Y: f32 [NB*NVOX][64].
// Block = 256 threads = 8 waves; block tile = 128 voxels x 64 out channels.
// Each wave owns 32 rows x 32 cols = 4 accumulator tiles, so every A and B
// fragment fetched is reused twice in registers (8 loads -> 8 WMMAs per tap).
// Nested dx/dy/dz loops hoist halo predicates and avoid per-tap divisions;
// dz (contiguous in memory) is fully unrolled.
// ---------------------------------------------------------------------------
__global__ __launch_bounds__(256)
void conv_wmma(const _Float16* __restrict__ A, const _Float16* __restrict__ W,
               float* __restrict__ Y) {
    const int lane    = threadIdx.x & 31;
    const int wave    = threadIdx.x >> 5;
    const int waveM   = wave & 3;     // 4 M-subtiles of 32 rows
    const int waveN   = wave >> 2;    // 2 N-subtiles of 32 cols
    const int rlane   = lane & 15;    // fragment row / column within 16-subtile
    const int halfsel = lane >> 4;    // K half-block selector

    const int tilesPerBatch = NVOX / 128;     // 864
    const int b    = blockIdx.x / tilesPerBatch;
    const int tile = blockIdx.x % tilesPerBatch;

    const int p0  = tile*128 + waveM*32 + rlane;   // lane's first A row
    const int p1  = p0 + 16;                       // lane's second A row
    const int px0 = p0 / 2304, py0 = (p0 / 48) % 48, pz0 = p0 % 48;
    const int px1 = p1 / 2304, py1 = (p1 / 48) % 48, pz1 = p1 % 48;
    const long rowBase0 = ((long)b * NVOX + p0) * CPAD;   // in halves
    const long rowBase1 = ((long)b * NVOX + p1) * CPAD;
    const int  n0 = waveN * 32;

    v8f acc00 = {}, acc01 = {}, acc10 = {}, acc11 = {};

    const _Float16* wt = W;   // advances dz-fastest: tap = (dx+2)*25+(dy+2)*5+(dz+2)
    for (int dx = -2; dx <= 2; ++dx) {
        const bool vx0 = (unsigned)(px0 + dx) < 48u;
        const bool vx1 = (unsigned)(px1 + dx) < 48u;
        for (int dy = -2; dy <= 2; ++dy) {
            const bool vy0 = vx0 && ((unsigned)(py0 + dy) < 48u);
            const bool vy1 = vx1 && ((unsigned)(py1 + dy) < 48u);
            const long offXY = (long)(dx*2304 + dy*48) * CPAD;

            if (dy < 2) {   // warm next dy-row into cache (global_prefetch_b8)
                const long offN = (long)(dx*2304 + (dy+1)*48) * CPAD;
                if (vy0) __builtin_prefetch(A + rowBase0 + offN, 0, 0);
            }

            #pragma unroll
            for (int dz = -2; dz <= 2; ++dz) {
                const bool v0 = vy0 && ((unsigned)(pz0 + dz) < 48u);
                const bool v1 = vy1 && ((unsigned)(pz1 + dz) < 48u);
                const long off = offXY + (long)dz * CPAD;
                #pragma unroll
                for (int ck = 0; ck < 2; ++ck) {
                    v16h a0 = {}, a1 = {};
                    if (v0) a0 = *(const v16h*)(A + rowBase0 + off + ck*32 + halfsel*16);
                    if (v1) a1 = *(const v16h*)(A + rowBase1 + off + ck*32 + halfsel*16);
                    const v16h b0 = *(const v16h*)(wt + (n0      + rlane)*64 + ck*32 + halfsel*16);
                    const v16h b1 = *(const v16h*)(wt + (n0 + 16 + rlane)*64 + ck*32 + halfsel*16);
                    acc00 = __builtin_amdgcn_wmma_f32_16x16x32_f16(false, a0, false, b0,
                                                                   (short)0, acc00, false, false);
                    acc01 = __builtin_amdgcn_wmma_f32_16x16x32_f16(false, a0, false, b1,
                                                                   (short)0, acc01, false, false);
                    acc10 = __builtin_amdgcn_wmma_f32_16x16x32_f16(false, a1, false, b0,
                                                                   (short)0, acc10, false, false);
                    acc11 = __builtin_amdgcn_wmma_f32_16x16x32_f16(false, a1, false, b1,
                                                                   (short)0, acc11, false, false);
                }
                wt += 64 * 64;
            }
        }
    }

    // D layout: VGPR r -> M = r + 8*halfsel, N = lane&15
    const long outBase = ((long)b*NVOX + (long)tile*128 + waveM*32) * CPAD;
    #pragma unroll
    for (int r = 0; r < 8; ++r) {
        const int row = r + 8*halfsel;
        Y[outBase + (long)row*CPAD        + n0      + rlane] = acc00[r];
        Y[outBase + (long)row*CPAD        + n0 + 16 + rlane] = acc01[r];
        Y[outBase + (long)(row+16)*CPAD   + n0      + rlane] = acc10[r];
        Y[outBase + (long)(row+16)*CPAD   + n0 + 16 + rlane] = acc11[r];
    }
}

// ---------------------------------------------------------------------------
// Elementwise stages
// ---------------------------------------------------------------------------
__global__ void pack_input(const float* __restrict__ x, _Float16* __restrict__ A) {
    const long i = (long)blockIdx.x * blockDim.x + threadIdx.x;
    if (i >= (long)NB * NVOX) return;
    _Float16* a = A + i * CPAD;
    #pragma unroll
    for (int c = 0; c < CPAD; ++c) a[c] = (_Float16)0.0f;
    a[0] = (_Float16)x[i];
}

__device__ __forceinline__ float gelu_tanh(float v) {
    return 0.5f * v * (1.0f + tanhf(0.7978845608f * (v + 0.044715f * v * v * v)));
}

__global__ void gate_kernel(const float* __restrict__ Y, _Float16* __restrict__ A) {
    const long i = (long)blockIdx.x * blockDim.x + threadIdx.x;
    if (i >= (long)NB * NVOX) return;
    const float* y = Y + i * CPAD;
    _Float16*    a = A + i * CPAD;
    #pragma unroll
    for (int c = 0; c < 10; ++c) a[c]      = (_Float16)gelu_tanh(y[c]);
    #pragma unroll
    for (int c = 0; c < 10; ++c) a[10 + c] = (_Float16)tanhf(y[10 + c]);
    float g[10];
    #pragma unroll
    for (int c = 0; c < 10; ++c) g[c] = 1.0f / (1.0f + expf(-y[20 + c]));
    #pragma unroll
    for (int c = 0; c < 30; ++c) a[20 + c] = (_Float16)(y[30 + c] * g[c / 3]);
    #pragma unroll
    for (int c = 50; c < CPAD; ++c) a[c] = (_Float16)0.0f;
}

__global__ void zero_out(float* __restrict__ out) {
    if (threadIdx.x < NB * 7) out[threadIdx.x] = 0.0f;
}

// Spatial sum of the 7 valid output channels per batch image.
// wave32 butterfly reduction, then one lane per wave hits LDS, then 7 global
// atomics per block.
__global__ void reduce_kernel(const float* __restrict__ Y, float* __restrict__ out) {
    __shared__ float sm[7];
    const int b = blockIdx.y;
    const long vox = (long)blockIdx.x * blockDim.x + threadIdx.x;
    const float* y = Y + ((long)b * NVOX + vox) * CPAD;
    float v[7];
    #pragma unroll
    for (int c = 0; c < 7; ++c) v[c] = y[c];
    #pragma unroll
    for (int s = 16; s > 0; s >>= 1)
        #pragma unroll
        for (int c = 0; c < 7; ++c) v[c] += __shfl_xor(v[c], s, 32);
    if (threadIdx.x < 7) sm[threadIdx.x] = 0.0f;
    __syncthreads();
    if ((threadIdx.x & 31) == 0)
        #pragma unroll
        for (int c = 0; c < 7; ++c) atomicAdd(&sm[c], v[c]);
    __syncthreads();
    if (threadIdx.x < 7) atomicAdd(&out[b*7 + threadIdx.x], sm[threadIdx.x]);
}

// ---------------------------------------------------------------------------
// Launch: pack -> 5x [build kernel, conv, gate] -> spatial-sum reduce.
// Workspace layout (bytes): A_half 28,311,552 | Y_f32 56,623,104 | W_T 1,024,000
// ---------------------------------------------------------------------------
extern "C" void kernel_launch(void* const* d_in, const int* in_sizes, int n_in,
                              void* d_out, int out_size, void* d_ws, size_t ws_size,
                              hipStream_t stream) {
    const float* x = (const float*)d_in[0];
    _Float16* A  = (_Float16*)d_ws;
    float*    Y  = (float*)((char*)d_ws + (size_t)NB*NVOX*CPAD*sizeof(_Float16));
    _Float16* Wt = (_Float16*)((char*)d_ws + (size_t)NB*NVOX*CPAD*sizeof(_Float16)
                                           + (size_t)NB*NVOX*CPAD*sizeof(float));

    const int nvoxAll = NB * NVOX;
    pack_input<<<(nvoxAll + 255)/256, 256, 0, stream>>>(x, A);

    for (int layer = 0; layer < 5; ++layer) {
        const float* tw = (const float*)d_in[1 + layer];
        const float* lw = (const float*)d_in[6 + layer];
        build_weights<<<NTAP, 64, 0, stream>>>(layer, tw, lw, Wt);
        conv_wmma<<<NB * (NVOX/128), 256, 0, stream>>>(A, Wt, Y);
        if (layer < 4)
            gate_kernel<<<(nvoxAll + 255)/256, 256, 0, stream>>>(Y, A);
    }

    zero_out<<<1, 32, 0, stream>>>((float*)d_out);
    reduce_kernel<<<dim3(NVOX/256, NB), 256, 0, stream>>>(Y, (float*)d_out);
}
